// IOC_52055003627642
// MI455X (gfx1250) — compile-verified, matrix operand order checked by reading, not run on hardware
//
#include <hip/hip_runtime.h>
#include <hip/hip_bf16.h>
#include <math.h>

// ---------------- Model constants ----------------
#define TD    30
#define NB    16
#define NAG   512          // Na = Nb * n
#define NPER  32           // agents per episode
#define NC    12
#define HID   48
#define NRING 6
#define NWEDG 6
#define NBIN  36           // 6*6
#define SPK   1728         // NBIN*HID
#define ROWS  6144         // Na*Nc
#define GRU3  144          // 3*HID
#define SCFK  96           // 32 + 16 + 48
#define HPAD  64           // HID padded to K%32==0
#define NREF  60           // 2*Td
#define RBR_F 0.34657359027997264f   // log(8)/6
#define PI_F  3.14159265358979323846f

typedef __bf16 bf16_t;
typedef __attribute__((ext_vector_type(16))) __bf16 v16bf;
typedef __attribute__((ext_vector_type(8)))  float  v8f;
typedef __attribute__((ext_vector_type(4)))  unsigned int u32x4;

union Frag { v16bf v; u32x4 q[2]; };

// ---------------- Utility kernels ----------------
__global__ void fill_f32(float* p, int n, float v) {
    int i = blockIdx.x * blockDim.x + threadIdx.x;
    if (i < n) p[i] = v;
}

// convert f32 weight [srcRows x srcCols] to bf16 [dstRows x dstLd] with zero pad
__global__ void cvt_pad_bf16(const float* __restrict__ src, bf16_t* __restrict__ dst,
                             int dstRows, int dstLd, int srcRows, int srcCols) {
    int i = blockIdx.x * blockDim.x + threadIdx.x;
    if (i >= dstRows * dstLd) return;
    int r = i / dstLd, c = i % dstLd;
    float v = (r < srcRows && c < srcCols) ? src[r * srcCols + c] : 0.f;
    dst[i] = (bf16_t)v;
}

// h0 broadcast: h[a*Nc+c][j] = Hx[a][j]; also bf16 padded copy with zero pad cols
__global__ void init_h_kernel(const float* __restrict__ Hx,
                              float* __restrict__ h, bf16_t* __restrict__ hbf) {
    int r = blockIdx.x;          // row = a*Nc + c
    int j = threadIdx.x;         // 0..63
    int a = r / NC;
    if (j < HID) {
        float v = Hx[a * HID + j];
        h[(size_t)r * HID + j]  = v;
        hbf[(size_t)r * HPAD + j] = (bf16_t)v;
    } else {
        hbf[(size_t)r * HPAD + j] = (bf16_t)0.f;
    }
}

// ---------------- Direct convolution (pad=2, 5x5) + ReLU ----------------
__global__ void conv2d_relu(const float* __restrict__ in, const float* __restrict__ w,
                            const float* __restrict__ bias, float* __restrict__ out,
                            float* __restrict__ padcopy,   // optional [N][Cout][34][34]
                            int N, int Cin, int Hin, int Win,
                            int Cout, int Hout, int Wout, int stride) {
    int idx = blockIdx.x * blockDim.x + threadIdx.x;
    int total = N * Cout * Hout * Wout;
    if (idx >= total) return;
    int ox = idx % Wout; int t = idx / Wout;
    int oy = t % Hout;   t /= Hout;
    int oc = t % Cout;   int nb = t / Cout;
    float acc = bias[oc];
    int iy0 = oy * stride - 2, ix0 = ox * stride - 2;
    for (int ic = 0; ic < Cin; ic++) {
        const float* ip = in + (size_t)(nb * Cin + ic) * Hin * Win;
        const float* wp = w + (size_t)(oc * Cin + ic) * 25;
        for (int ky = 0; ky < 5; ky++) {
            int iy = iy0 + ky;
            if (iy < 0 || iy >= Hin) continue;
            for (int kx = 0; kx < 5; kx++) {
                int ix = ix0 + kx;
                if (ix < 0 || ix >= Win) continue;
                acc += ip[iy * Win + ix] * wp[ky * 5 + kx];
            }
        }
    }
    acc = fmaxf(acc, 0.f);
    out[idx] = acc;
    if (padcopy)
        padcopy[((size_t)(nb * Cout + oc) * 34 + (oy + 1)) * 34 + (ox + 1)] = acc;
}

// ---------------- WMMA bf16 GEMM: C[M x N] = A[M x K] * B^T (B stored [N x K]) ----------------
// One wave computes a 16 x (NT*16) strip: A fragment loaded once per K-step, reused
// across NT WMMAs (register-level A reuse -> A traffic / NT).
// grid.x = M/16 tiles, grid.y = N/(NT*16) groups, block = 32 (one wave)
template <int NT>
__global__ void wmma_gemm_bf16(const bf16_t* __restrict__ A, int lda,
                               const bf16_t* __restrict__ B, int ldb,
                               float* __restrict__ C, int ldc,
                               int K, const float* __restrict__ bias,
                               int relu, int nvalid) {
    int lane = threadIdx.x & 31;
    int row  = lane & 15;
    int koff = (lane < 16) ? 0 : 8;              // 16-bit A/B fragment layout (ISA 7.12.2)
    const bf16_t* Ar = A + (size_t)(blockIdx.x * 16 + row) * lda;
    const bf16_t* Br = B + (size_t)((blockIdx.y * NT) * 16 + row) * ldb;
    const v8f zero = {};
    v8f acc[NT];
#pragma unroll
    for (int nt = 0; nt < NT; nt++) acc[nt] = zero;

    for (int k0 = 0; k0 < K; k0 += 32) {
        Frag a;
        a.q[0] = *(const u32x4*)(Ar + k0 + koff);        // K = koff..koff+7
        a.q[1] = *(const u32x4*)(Ar + k0 + koff + 16);   // K = koff+16..koff+23
        if (k0 + 512 < K)                                // long-K (fcsp): prefetch A stream
            __builtin_prefetch(Ar + k0 + 512, 0, 0);
#pragma unroll
        for (int nt = 0; nt < NT; nt++) {
            Frag b;
            const bf16_t* Bt = Br + (size_t)nt * 16 * ldb;
            b.q[0] = *(const u32x4*)(Bt + k0 + koff);
            b.q[1] = *(const u32x4*)(Bt + k0 + koff + 16);
            acc[nt] = __builtin_amdgcn_wmma_f32_16x16x32_bf16(
                /*neg_a=*/false, a.v, /*neg_b=*/false, b.v,
                /*c_mod=*/(short)0, acc[nt], /*reuse_a=*/false, /*reuse_b=*/false);
        }
    }

    int mbase = blockIdx.x * 16 + ((lane >= 16) ? 8 : 0);  // C layout: VGPR v -> M=v or v+8
#pragma unroll
    for (int nt = 0; nt < NT; nt++) {
        int ncol = (blockIdx.y * NT + nt) * 16 + (lane & 15);
        if (ncol >= nvalid) continue;
        float bv = bias ? bias[ncol] : 0.f;
#pragma unroll
        for (int v = 0; v < 8; v++) {
            float val = acc[nt][v] + bv;
            if (relu) val = fmaxf(val, 0.f);
            C[(size_t)(mbase + v) * ldc + ncol] = val;
        }
    }
}

// ---------------- Social pooling: build sp row [1728] as bf16 ----------------
// grid = ROWS, block = 64
__global__ void social_pool_kernel(const float* __restrict__ position, int t,
                                   const float* __restrict__ h,
                                   bf16_t* __restrict__ spbuf) {
    int r = blockIdx.x;
    int a = r / NC, c = r % NC;
    int e = a >> 5;             // episode (n=32)
    int ego = a & 31;
    int tid = threadIdx.x;

    __shared__ float px[NPER], py[NPER];
    __shared__ int   binS[NPER];
    __shared__ int   cnt[NBIN];
    __shared__ float hsh[NPER * HID];
    __shared__ float sp[SPK];

    if (tid < NPER) {
        int ag = e * NPER + tid;
        const float* p = position + (((size_t)t * NAG + ag) * NC + c) * 2;
        px[tid] = p[0]; py[tid] = p[1];
    }
    if (tid < NBIN) cnt[tid] = 0;
    __syncthreads();

    if (tid < NPER) {
        float xd = px[ego] - px[tid];
        float yd = py[ego] - py[tid];
        float dist = sqrtf(xd * xd + yd * yd);
        int ring = (int)floorf(logf(dist * 2.0f + 1e-6f) / RBR_F);   // RMIN = 0.5
        int b = -1;
        if (ring >= 0 && ring < NRING) {
            float ang = atan2f(yd, xd) + PI_F - 1e-6f;
            int wdg = (int)floorf(ang / (2.f * PI_F) * (float)NWEDG);
            wdg = min(max(wdg, 0), NWEDG - 1);
            b = ring * NWEDG + wdg;
            atomicAdd(&cnt[b], 1);
        }
        binS[tid] = b;
    }
    for (int i = tid; i < NPER * HID; i += 64) {
        int s = i / HID, j = i % HID;
        hsh[i] = h[((size_t)(e * NPER + s) * NC + c) * HID + j];
    }
    for (int i = tid; i < SPK; i += 64) sp[i] = 0.f;
    __syncthreads();

    if (tid < HID) {                         // each thread owns one hid dim column
        for (int s = 0; s < NPER; s++) {
            int b = binS[s];
            if (b >= 0) sp[b * HID + tid] += hsh[s * HID + tid];
        }
    }
    __syncthreads();

    bf16_t* out = spbuf + (size_t)r * SPK;
    for (int i = tid; i < SPK; i += 64) {
        float cd = (float)max(cnt[i / HID], 1);
        out[i] = (bf16_t)(sp[i] / cd);
    }
}

// ---------------- Fused scene-gather + fcvel + sp_enc concat -> scf bf16 [ROWS][96] ----------
// grid = ROWS, block = 96
__global__ void concat_scf_kernel(const float* __restrict__ position,
                                  const float* __restrict__ velocity, int t,
                                  const float* __restrict__ fmp,       // [NB][32][34][34]
                                  const float* __restrict__ fcvel_w,   // [16][2]
                                  const float* __restrict__ fcvel_b,
                                  const float* __restrict__ sp_enc,    // [ROWS][48]
                                  bf16_t* __restrict__ scf) {
    int r = blockIdx.x;
    int j = threadIdx.x;
    int a = r / NC, c = r % NC, e = a >> 5;
    float outv;
    if (j < 32) {
        const float* p = position + (((size_t)t * NAG + a) * NC + c) * 2;
        float mx = p[0] * (32.f / 112.f) + 17.f;   // (p+56)*32/112 + 1
        float my = p[1] * (32.f / 112.f) + 17.f;
        int xi = min(max((int)floorf(mx), 0), 33);
        int yi = min(max((int)floorf(my), 0), 33);
        outv = fmp[((size_t)(e * 32 + j) * 34 + yi) * 34 + xi];
    } else if (j < 48) {
        int k = j - 32;
        const float* v = velocity + (((size_t)t * NAG + a) * NC + c) * 2;
        outv = fmaxf(v[0] * fcvel_w[k * 2] + v[1] * fcvel_w[k * 2 + 1] + fcvel_b[k], 0.f);
    } else {
        outv = sp_enc[(size_t)r * HID + (j - 48)];
    }
    scf[(size_t)r * SCFK + j] = (bf16_t)outv;
}

// ---------------- GRU elementwise + score head ----------------
// grid = ROWS, block = 48; biases already folded into gi/gh by the GEMMs
__global__ void gru_kernel(const float* __restrict__ gi, const float* __restrict__ gh,
                           float* __restrict__ h, bf16_t* __restrict__ hbf,
                           const float* __restrict__ score_w, const float* __restrict__ score_b,
                           float* __restrict__ scores_t) {
    int r = blockIdx.x, j = threadIdx.x;
    const float* gir = gi + (size_t)r * GRU3;
    const float* ghr = gh + (size_t)r * GRU3;
    float rg = 1.f / (1.f + expf(-(gir[j] + ghr[j])));
    float zg = 1.f / (1.f + expf(-(gir[HID + j] + ghr[HID + j])));
    float nn = tanhf(gir[2 * HID + j] + rg * ghr[2 * HID + j]);
    float hprev = h[(size_t)r * HID + j];
    float h2 = (1.f - zg) * nn + zg * hprev;
    h[(size_t)r * HID + j]   = h2;
    hbf[(size_t)r * HPAD + j] = (bf16_t)h2;

    __shared__ float sh[HID];
    sh[j] = h2 * score_w[j];
    __syncthreads();
    if (j == 0) {
        float s = score_b[0];
        for (int k = 0; k < HID; k++) s += sh[k];
        scores_t[r] = fmaxf(s, 0.f);
    }
}

// ---------------- refine output scatter into dY [Td,Na,Nc,2] ----------------
__global__ void refine_scatter(const float* __restrict__ refout, float* __restrict__ dY) {
    int i = blockIdx.x * blockDim.x + threadIdx.x;
    if (i >= ROWS * NREF) return;
    int r = i / NREF, col = i % NREF;
    int t = col >> 1, d = col & 1;
    dY[((size_t)t * ROWS + r) * 2 + d] = refout[i];
}

// ---------------- Host launch ----------------
extern "C" void kernel_launch(void* const* d_in, const int* in_sizes, int n_in,
                              void* d_out, int out_size, void* d_ws, size_t ws_size,
                              hipStream_t stream) {
    (void)in_sizes; (void)n_in; (void)out_size; (void)ws_size;
    const float* velocity = (const float*)d_in[0];
    const float* position = (const float*)d_in[1];
    const float* Hx       = (const float*)d_in[2];
    const float* scene    = (const float*)d_in[3];
    // d_in[4] num_tgt_trajs (all == 32, unused)
    const float* conv1_w = (const float*)d_in[5];
    const float* conv1_b = (const float*)d_in[6];
    const float* conv2_w = (const float*)d_in[7];
    const float* conv2_b = (const float*)d_in[8];
    const float* conv3_w = (const float*)d_in[9];
    const float* conv3_b = (const float*)d_in[10];
    const float* fcvel_w = (const float*)d_in[11];
    const float* fcvel_b = (const float*)d_in[12];
    const float* fcsp_w  = (const float*)d_in[13];
    const float* fcsp_b  = (const float*)d_in[14];
    const float* gru_wih = (const float*)d_in[15];
    const float* gru_whh = (const float*)d_in[16];
    const float* gru_bih = (const float*)d_in[17];
    const float* gru_bhh = (const float*)d_in[18];
    const float* score_w = (const float*)d_in[19];
    const float* score_b = (const float*)d_in[20];
    const float* refine_w = (const float*)d_in[21];
    const float* refine_b = (const float*)d_in[22];

    float* out = (float*)d_out;
    float* out_scores = out;                      // [Td, Na, Nc, 1]
    float* out_dY     = out + (size_t)TD * ROWS;  // [Td, Na, Nc, 2]
    float* out_feat   = out + (size_t)TD * ROWS * 3; // [Nb, 32, 32, 32]

    // workspace carve-up
    char* base = (char*)d_ws;
    size_t off = 0;
    auto alloc = [&](size_t bytes) { char* p = base + off; off += (bytes + 255) & ~(size_t)255; return p; };
    float*  x1       = (float*)alloc((size_t)NB * 16 * 32 * 32 * 4);
    float*  x2       = (float*)alloc((size_t)NB * 32 * 32 * 32 * 4);
    float*  fmp      = (float*)alloc((size_t)NB * 32 * 34 * 34 * 4);
    bf16_t* spbuf    = (bf16_t*)alloc((size_t)ROWS * SPK * 2);
    float*  sp_enc   = (float*)alloc((size_t)ROWS * HID * 4);
    bf16_t* scf      = (bf16_t*)alloc((size_t)ROWS * SCFK * 2);
    float*  gi       = (float*)alloc((size_t)ROWS * GRU3 * 4);
    float*  gh       = (float*)alloc((size_t)ROWS * GRU3 * 4);
    float*  hbuf     = (float*)alloc((size_t)ROWS * HID * 4);
    bf16_t* hbf      = (bf16_t*)alloc((size_t)ROWS * HPAD * 2);
    bf16_t* w_fcsp   = (bf16_t*)alloc((size_t)HID * SPK * 2);
    bf16_t* w_wih    = (bf16_t*)alloc((size_t)GRU3 * SCFK * 2);
    bf16_t* w_whh    = (bf16_t*)alloc((size_t)GRU3 * HPAD * 2);
    bf16_t* w_ref    = (bf16_t*)alloc((size_t)64 * HPAD * 2);
    float*  refout   = (float*)alloc((size_t)ROWS * NREF * 4);

    // ---- setup: weight conversion (bf16, zero-padded K), h init, fmp zero ----
    {
        int n = HID * SPK;
        cvt_pad_bf16<<<(n + 255) / 256, 256, 0, stream>>>(fcsp_w, w_fcsp, HID, SPK, HID, SPK);
        n = GRU3 * SCFK;
        cvt_pad_bf16<<<(n + 255) / 256, 256, 0, stream>>>(gru_wih, w_wih, GRU3, SCFK, GRU3, SCFK);
        n = GRU3 * HPAD;
        cvt_pad_bf16<<<(n + 255) / 256, 256, 0, stream>>>(gru_whh, w_whh, GRU3, HPAD, GRU3, HID);
        n = 64 * HPAD;
        cvt_pad_bf16<<<(n + 255) / 256, 256, 0, stream>>>(refine_w, w_ref, 64, HPAD, NREF, HID);
        init_h_kernel<<<ROWS, HPAD, 0, stream>>>(Hx, hbuf, hbf);
        int nf = NB * 32 * 34 * 34;
        fill_f32<<<(nf + 255) / 256, 256, 0, stream>>>(fmp, nf, 0.f);
    }

    // ---- CNN ----
    {
        int t1 = NB * 16 * 32 * 32;
        conv2d_relu<<<(t1 + 255) / 256, 256, 0, stream>>>(scene, conv1_w, conv1_b, x1, nullptr,
                                                          NB, 3, 64, 64, 16, 32, 32, 2);
        int t2 = NB * 32 * 32 * 32;
        conv2d_relu<<<(t2 + 255) / 256, 256, 0, stream>>>(x1, conv2_w, conv2_b, x2, nullptr,
                                                          NB, 16, 32, 32, 32, 32, 32, 1);
        conv2d_relu<<<(t2 + 255) / 256, 256, 0, stream>>>(x2, conv3_w, conv3_b, out_feat, fmp,
                                                          NB, 32, 32, 32, 32, 32, 32, 1);
    }

    // ---- recurrent loop ----
    for (int t = 0; t < TD; t++) {
        social_pool_kernel<<<ROWS, 64, 0, stream>>>(position, t, hbuf, spbuf);
        // sp_enc = relu(sp @ fcsp_w^T + b)   M=6144 K=1728 N=48 (3 N-tiles per wave)
        wmma_gemm_bf16<3><<<dim3(ROWS / 16, 1), 32, 0, stream>>>(
            spbuf, SPK, w_fcsp, SPK, sp_enc, HID, SPK, fcsp_b, 1, HID);
        concat_scf_kernel<<<ROWS, SCFK, 0, stream>>>(position, velocity, t, fmp,
                                                     fcvel_b ? fcvel_w : fcvel_w, fcvel_b, sp_enc, scf);
        // gi = scf @ wih^T + bih   M=6144 K=96 N=144 (9 N-tiles per wave)
        wmma_gemm_bf16<9><<<dim3(ROWS / 16, 1), 32, 0, stream>>>(
            scf, SCFK, w_wih, SCFK, gi, GRU3, SCFK, gru_bih, 0, GRU3);
        // gh = h @ whh^T + bhh     M=6144 K=64(pad) N=144
        wmma_gemm_bf16<9><<<dim3(ROWS / 16, 1), 32, 0, stream>>>(
            hbf, HPAD, w_whh, HPAD, gh, GRU3, HPAD, gru_bhh, 0, GRU3);
        gru_kernel<<<ROWS, HID, 0, stream>>>(gi, gh, hbuf, hbf, score_w, score_b,
                                             out_scores + (size_t)t * ROWS);
    }

    // ---- refine head: dY = hN @ refine_w^T + b   M=6144 K=64(pad) N=60 (4 tiles, mask to 60) ----
    wmma_gemm_bf16<4><<<dim3(ROWS / 16, 1), 32, 0, stream>>>(
        hbf, HPAD, w_ref, HPAD, refout, NREF, HPAD, refine_b, 0, NREF);
    {
        int n = ROWS * NREF;
        refine_scatter<<<(n + 255) / 256, 256, 0, stream>>>(refout, out_dY);
    }
}
